// MoETransformerBlock_73512660238759
// MI455X (gfx1250) — compile-verified, matrix operand order checked by reading, not run on hardware
//
#include <hip/hip_runtime.h>

// ---------------------------------------------------------------------------
// Problem dims: B=2,S=2048,D=1024,H=16,HD=64,F=4096,E=8,K=2
// ---------------------------------------------------------------------------
#define BB 2
#define SS 2048
#define DD 1024
#define HH 16
#define HDHD 64
#define FF 4096
#define EE 8
#define TT (BB * SS)   // 4096 tokens

typedef __attribute__((ext_vector_type(16))) __bf16 v16bf;
typedef __attribute__((ext_vector_type(8)))  float  v8f;

#define WMMA_BF(a, b, c) \
  __builtin_amdgcn_wmma_f32_16x16x32_bf16(false, (a), false, (b), (short)0, (c), false, false)

// ---------------------------------------------------------------------------
// Tensor Data Mover: strided 2D tile -> LDS. D# per ISA ch.8 (CDNA5 ISA).
// Toolchain probe: builtin takes 6 args (g0 u32x4, g1 i32x8, i32x4, i32x4,
// i32x8, i32 cpol).
// ---------------------------------------------------------------------------
#if __has_builtin(__builtin_amdgcn_tensor_load_to_lds) && \
    __has_builtin(__builtin_amdgcn_s_wait_tensorcnt)
#define HAVE_TDM 1
typedef __attribute__((ext_vector_type(4))) unsigned tdm_u32x4;
typedef __attribute__((ext_vector_type(8))) int      tdm_i32x8;
typedef __attribute__((ext_vector_type(4))) int      tdm_i32x4;

// Load tile_d1 lines of tile_d0 contiguous bf16 elements (line stride
// stride0 elements) from gptr into contiguous LDS at lds_off.
__device__ __forceinline__ void tdm_load_2d(const void* gptr, unsigned lds_off,
                                            unsigned tensor_d0, unsigned tensor_d1,
                                            unsigned tile_d0, unsigned tile_d1,
                                            unsigned stride0) {
  unsigned long long ga = (unsigned long long)gptr;
  tdm_u32x4 g0;
  g0[0] = 1u;                                               // count=1 (valid D#)
  g0[1] = lds_off;                                          // lds_addr (bytes)
  g0[2] = (unsigned)ga;                                     // global_addr[31:0]
  g0[3] = (unsigned)((ga >> 32) & 0x1FFFFFFu) | (2u << 30); // addr[56:32] | type=2
  tdm_i32x8 g1;
  g1[0] = (int)(1u << 16);                                  // data_size=2B; mask=0
  g1[1] = (int)((tensor_d0 & 0xFFFFu) << 16);               // tensor_dim0 lo
  g1[2] = (int)((tensor_d0 >> 16) | ((tensor_d1 & 0xFFFFu) << 16));
  g1[3] = (int)((tensor_d1 >> 16) | (tile_d0 << 16));       // tile_dim0
  g1[4] = (int)tile_d1;                                     // tile_dim1 (tile_dim2=0)
  g1[5] = (int)stride0;                                     // tensor_dim0_stride lo
  g1[6] = 0;
  g1[7] = 0;
  tdm_i32x4 z4 = {0, 0, 0, 0};
  tdm_i32x8 z8 = {0, 0, 0, 0, 0, 0, 0, 0};
  __builtin_amdgcn_tensor_load_to_lds(g0, g1, z4, z4, z8, 0);
}
__device__ __forceinline__ unsigned lds_off_of(const void* p) {
  return (unsigned)(unsigned long long)p;   // low 32 bits of generic LDS ptr
}
#endif

// ---------------------------------------------------------------------------
// bf16 helpers (bit-level, RNE)
// ---------------------------------------------------------------------------
__device__ __forceinline__ __bf16 f2bf(float f) {
  union { float f; unsigned u; } v; v.f = f;
  unsigned r = (v.u + 0x7FFFu + ((v.u >> 16) & 1u)) >> 16;
  unsigned short us = (unsigned short)r;
  return __builtin_bit_cast(__bf16, us);
}
__device__ __forceinline__ float bf2f(__bf16 b) {
  unsigned short us = __builtin_bit_cast(unsigned short, b);
  union { unsigned u; float f; } v; v.u = ((unsigned)us) << 16;
  return v.f;
}

// ---------------------------------------------------------------------------
// CDNA5 16x16x32 bf16 fragment K-index mapping (ISA 7.12.2). Per lane the 16
// elements form two contiguous 16B runs -> 2x ds_load_b128.
// ---------------------------------------------------------------------------
__device__ __forceinline__ int frag_koff(int e, int lane) {
  int r = e >> 1, s = e & 1;
  int base = (r < 4) ? (2 * r + s) : (16 + 2 * (r - 4) + s);
  return base + ((lane >> 4) & 1) * 8;
}

__device__ __forceinline__ v16bf lds_frag(const __bf16* s, int row0, int ldk, int kshift) {
  int lane = threadIdx.x & 31;
  int m = lane & 15;
  v16bf a;
#pragma unroll
  for (int e = 0; e < 16; ++e) a[e] = s[(row0 + m) * ldk + kshift + frag_koff(e, lane)];
  return a;
}

// ---------------------------------------------------------------------------
// Transpose-convert: out_bf16[z][n][k] = f2bf(in_f32[z][k][n])
// ---------------------------------------------------------------------------
__launch_bounds__(256)
__global__ void cvt_t_bf16_kernel(const float* __restrict__ in, __bf16* __restrict__ out,
                                  int K, int N) {
  __shared__ float tile[32][33];
  const float* inz = in + (size_t)blockIdx.z * K * N;
  __bf16* outz = out + (size_t)blockIdx.z * K * N;
  int k0 = blockIdx.y * 32, n0 = blockIdx.x * 32;
  int tx = threadIdx.x & 31, ty = threadIdx.x >> 5;
  for (int r = ty; r < 32; r += 8)
    tile[r][tx] = inz[(size_t)(k0 + r) * N + n0 + tx];
  __syncthreads();
  for (int r = ty; r < 32; r += 8)
    outz[(size_t)(n0 + r) * K + k0 + tx] = f2bf(tile[tx][r]);
}

// ---------------------------------------------------------------------------
// RMSNorm (one block per row), bf16 output
// ---------------------------------------------------------------------------
__launch_bounds__(256)
__global__ void rmsnorm_kernel(const float* __restrict__ x, const float* __restrict__ w,
                               __bf16* __restrict__ out, int D_) {
  int row = blockIdx.x;
  const float* xr = x + (size_t)row * D_;
  float ss = 0.f;
  for (int d = threadIdx.x; d < D_; d += 256) { float v = xr[d]; ss += v * v; }
  __shared__ float red[256];
  red[threadIdx.x] = ss;
  __syncthreads();
  for (int s = 128; s > 0; s >>= 1) {
    if (threadIdx.x < s) red[threadIdx.x] += red[threadIdx.x + s];
    __syncthreads();
  }
  float inv = rsqrtf(red[0] / (float)D_ + 1e-5f);
  for (int d = threadIdx.x; d < D_; d += 256)
    out[(size_t)row * D_ + d] = f2bf(xr[d] * inv * w[d]);
}

// ---------------------------------------------------------------------------
// Tiled WMMA GEMM: C[M,N] = A[M,K] @ B^T, BwT pre-transposed [N][K].
// 256 thr / 8 waves, tile 128x64, K-step 32. B tile staged by TDM.
// out_mode: 0 fp32; 1 bf16; 2 bf16 V-transposed [B][H][HD][S].
// ---------------------------------------------------------------------------
__launch_bounds__(256)
__global__ void gemm_bf16_kernel(const __bf16* __restrict__ A, const __bf16* __restrict__ BwT,
                                 void* __restrict__ C, int M, int N, int Kd, int out_mode) {
  __shared__ __align__(16) __bf16 sA[128 * 32];
  __shared__ __align__(16) __bf16 sB[64 * 32];   // [n][k]
  int bm = blockIdx.y * 128;
  int bn = blockIdx.x * 64;
  int tid = threadIdx.x, wv = tid >> 5, lane = tid & 31;
  int hi = (lane >> 4) & 1, ln = lane & 15;
  v8f acc[4] = {};
  for (int k0 = 0; k0 < Kd; k0 += 32) {
#ifdef HAVE_TDM
    if (wv == 0)
      tdm_load_2d(BwT + (size_t)bn * Kd + k0, lds_off_of(sB),
                  (unsigned)Kd, (unsigned)N, 32u, 64u, (unsigned)Kd);
#else
    for (int i = tid; i < 256; i += 256) {
      int r = i >> 2, c = (i & 3) << 3;
      ((uint4*)sB)[i] = *(const uint4*)(BwT + (size_t)(bn + r) * Kd + k0 + c);
    }
#endif
    {   // A: two-phase copy (both loads issued, then both stores)
      int i0 = tid, i1 = tid + 256;
      const uint4 a0 = *(const uint4*)(A + (size_t)(bm + (i0 >> 2)) * Kd + k0 + ((i0 & 3) << 3));
      const uint4 a1 = *(const uint4*)(A + (size_t)(bm + (i1 >> 2)) * Kd + k0 + ((i1 & 3) << 3));
      ((uint4*)sA)[i0] = a0;
      ((uint4*)sA)[i1] = a1;
    }
    if (k0 + 32 < Kd)
      __builtin_prefetch(A + (size_t)(bm + (tid & 127)) * Kd + k0 + 32, 0, 1);
#ifdef HAVE_TDM
    if (wv == 0) __builtin_amdgcn_s_wait_tensorcnt(0);
#endif
    __syncthreads();
    v16bf af = lds_frag(sA, wv * 16, 32, 0);
#pragma unroll
    for (int nt = 0; nt < 4; ++nt) {
      v16bf bf = lds_frag(sB, nt * 16, 32, 0);
      acc[nt] = WMMA_BF(af, bf, acc[nt]);
    }
    __syncthreads();
  }
#pragma unroll
  for (int nt = 0; nt < 4; ++nt) {
#pragma unroll
    for (int r = 0; r < 8; ++r) {
      size_t m = bm + wv * 16 + r + hi * 8;
      size_t n = bn + nt * 16 + ln;
      if (out_mode == 0) {
        ((float*)C)[m * N + n] = acc[nt][r];
      } else if (out_mode == 1) {
        ((__bf16*)C)[m * N + n] = f2bf(acc[nt][r]);
      } else {
        size_t bbi = m / SS, si = m % SS;
        size_t hhh = n / HDHD, ddd = n % HDHD;
        ((__bf16*)C)[((bbi * HH + hhh) * HDHD + ddd) * SS + si] = f2bf(acc[nt][r]);
      }
    }
  }
}

// ---------------------------------------------------------------------------
// RoPE applied in-place to q and k (bf16)
// ---------------------------------------------------------------------------
__global__ void rope_kernel(__bf16* __restrict__ q, __bf16* __restrict__ k,
                            const int* __restrict__ pos, int T, int H_, int HD_) {
  int half = HD_ >> 1;
  int total = T * H_ * half;
  int idx = blockIdx.x * blockDim.x + threadIdx.x;
  if (idx >= total) return;
  int i = idx % half;
  int th = idx / half;
  int hh = th % H_;
  int t = th / H_;
  float p = (float)pos[t];
  float inv_freq = __powf(10000.0f, -(float)i / (float)half);
  float ang = p * inv_freq;
  float sv, cv;
  __sincosf(ang, &sv, &cv);
  size_t base = (size_t)t * (H_ * HD_) + (size_t)hh * HD_ + 2 * i;
  float q1 = bf2f(q[base]), q2 = bf2f(q[base + 1]);
  q[base]     = f2bf(q1 * cv - q2 * sv);
  q[base + 1] = f2bf(q1 * sv + q2 * cv);
  float k1 = bf2f(k[base]), k2 = bf2f(k[base + 1]);
  k[base]     = f2bf(k1 * cv - k2 * sv);
  k[base + 1] = f2bf(k1 * sv + k2 * cv);
}

// ---------------------------------------------------------------------------
// Flash attention, causal. 128 thr = 4 waves on 64 q-rows of one (b,h).
// K / V^T tiles staged by TDM; P staged via per-wave LDS.
// ---------------------------------------------------------------------------
__launch_bounds__(128)
__global__ void attn_kernel(const __bf16* __restrict__ Qm, const __bf16* __restrict__ Km,
                            const __bf16* __restrict__ vT, __bf16* __restrict__ Om,
                            int S_, int H_, int HD_) {
  const int D_ = H_ * HD_;
  int bh = blockIdx.x;
  int bb = bh / H_, hh = bh % H_;
  int wv = threadIdx.x >> 5, lane = threadIdx.x & 31;
  int hi = (lane >> 4) & 1, ln = lane & 15;
  int qblk = blockIdx.y * 64;
  int q0 = qblk + wv * 16;

  __shared__ __align__(16) __bf16 sK[32 * 64];     // [key][d]
  __shared__ __align__(16) __bf16 sV[64 * 32];     // [d][key]
  __shared__ __align__(16) __bf16 sP[4][16 * 32];  // per-wave P tile
  __bf16* myP = sP[wv];

  size_t rowbase = (size_t)bb * S_ * D_ + (size_t)hh * HD_;
  const __bf16* vTh = vT + (size_t)(bb * H_ + hh) * HD_ * S_;

  v16bf aq0, aq1;
#pragma unroll
  for (int e = 0; e < 16; ++e) {
    int d = frag_koff(e, lane);
    aq0[e] = Qm[rowbase + (size_t)(q0 + ln) * D_ + d];
    aq1[e] = Qm[rowbase + (size_t)(q0 + ln) * D_ + 32 + d];
  }

  float rmax[8], rsum[8];
#pragma unroll
  for (int r = 0; r < 8; ++r) { rmax[r] = -1e30f; rsum[r] = 0.f; }
  v8f o[4] = {};

  int kend = q0 + 16;
  int kend_blk = qblk + 64;
  for (int k0 = 0; k0 < kend_blk; k0 += 32) {
#ifdef HAVE_TDM
    if (wv == 0) {
      tdm_load_2d(Km + rowbase + (size_t)k0 * D_, lds_off_of(sK),
                  (unsigned)D_, (unsigned)S_, 64u, 32u, (unsigned)D_);
      tdm_load_2d(vTh + k0, lds_off_of(sV),
                  (unsigned)S_, (unsigned)HD_, 32u, 64u, (unsigned)S_);
      __builtin_amdgcn_s_wait_tensorcnt(0);
    }
#else
    {
      int i0 = threadIdx.x, i1 = threadIdx.x + 128;
      const uint4 ka = *(const uint4*)(Km + rowbase + (size_t)(k0 + (i0 >> 3)) * D_ + ((i0 & 7) << 3));
      const uint4 kb = *(const uint4*)(Km + rowbase + (size_t)(k0 + (i1 >> 3)) * D_ + ((i1 & 7) << 3));
      const uint4 va = *(const uint4*)(vTh + (size_t)(i0 >> 2) * S_ + k0 + ((i0 & 3) << 3));
      const uint4 vb = *(const uint4*)(vTh + (size_t)(i1 >> 2) * S_ + k0 + ((i1 & 3) << 3));
      ((uint4*)sK)[i0] = ka;
      ((uint4*)sK)[i1] = kb;
      ((uint4*)sV)[i0] = va;
      ((uint4*)sV)[i1] = vb;
    }
#endif
    __syncthreads();

    if (k0 < kend) {   // wave-uniform: skip fully-masked steps
      float sc[2][8];
#pragma unroll
      for (int hf = 0; hf < 2; ++hf) {
        int kb = k0 + hf * 16;
        v16bf bk0 = lds_frag(sK, hf * 16, 64, 0);
        v16bf bk1 = lds_frag(sK, hf * 16, 64, 32);
        v8f s8 = {};
        s8 = WMMA_BF(aq0, bk0, s8);
        s8 = WMMA_BF(aq1, bk1, s8);
#pragma unroll
        for (int r = 0; r < 8; ++r) {
          int m = q0 + r + hi * 8;
          int kidx = kb + ln;
          float svv = s8[r] * 0.125f;
          if (kidx > m) svv = -1e30f;
          sc[hf][r] = svv;
        }
      }
#pragma unroll
      for (int r = 0; r < 8; ++r) {
        float mloc = fmaxf(sc[0][r], sc[1][r]);
        for (int off = 8; off >= 1; off >>= 1) mloc = fmaxf(mloc, __shfl_xor(mloc, off, 32));
        float mnew = fmaxf(rmax[r], mloc);
        float scale = __expf(rmax[r] - mnew);
        float p0 = __expf(sc[0][r] - mnew);
        float p1 = __expf(sc[1][r] - mnew);
        float ps = p0 + p1;
        for (int off = 8; off >= 1; off >>= 1) ps += __shfl_xor(ps, off, 32);
        rsum[r] = rsum[r] * scale + ps;
        rmax[r] = mnew;
        o[0][r] *= scale; o[1][r] *= scale; o[2][r] *= scale; o[3][r] *= scale;
        myP[(r + hi * 8) * 32 + ln]      = f2bf(p0);
        myP[(r + hi * 8) * 32 + 16 + ln] = f2bf(p1);
      }
      asm volatile("s_wait_dscnt 0" ::: "memory");   // wave-local LDS RAW
      v16bf ap = lds_frag(myP, 0, 32, 0);
#pragma unroll
      for (int nt = 0; nt < 4; ++nt) {
        v16bf bv = lds_frag(sV, nt * 16, 32, 0);
        o[nt] = WMMA_BF(ap, bv, o[nt]);
      }
    }
    __syncthreads();
  }
#pragma unroll
  for (int r = 0; r < 8; ++r) {
    float inv = 1.0f / rsum[r];
    size_t obase = rowbase + (size_t)(q0 + r + hi * 8) * D_;
    Om[obase + ln]      = f2bf(o[0][r] * inv);
    Om[obase + 16 + ln] = f2bf(o[1][r] * inv);
    Om[obase + 32 + ln] = f2bf(o[2][r] * inv);
    Om[obase + 48 + ln] = f2bf(o[3][r] * inv);
  }
}

// ---------------------------------------------------------------------------
// elementwise add
// ---------------------------------------------------------------------------
__global__ void add2_kernel(const float* __restrict__ a, const float* __restrict__ b,
                            float* __restrict__ out, size_t n) {
  size_t i = (size_t)blockIdx.x * blockDim.x + threadIdx.x;
  if (i < n) out[i] = a[i] + b[i];
}

// ---------------------------------------------------------------------------
// Gate: logits = xn @ gate_w, top-2, renormalize, build expert lists
// ---------------------------------------------------------------------------
__launch_bounds__(256)
__global__ void gate_topk_kernel(const __bf16* __restrict__ xn, const float* __restrict__ gw,
                                 int* __restrict__ counts, int* __restrict__ tok_list,
                                 float* __restrict__ wt_list, int T, int D_) {
  int t = blockIdx.x;
  float acc[EE] = {};
  for (int d = threadIdx.x; d < D_; d += 256) {
    float xv = bf2f(xn[(size_t)t * D_ + d]);
#pragma unroll
    for (int e = 0; e < EE; ++e) acc[e] += xv * gw[d * EE + e];
  }
  __shared__ float red[256];
  __shared__ float logits[EE];
  for (int e = 0; e < EE; ++e) {
    red[threadIdx.x] = acc[e];
    __syncthreads();
    for (int s = 128; s > 0; s >>= 1) {
      if (threadIdx.x < s) red[threadIdx.x] += red[threadIdx.x + s];
      __syncthreads();
    }
    if (threadIdx.x == 0) logits[e] = red[0];
    __syncthreads();
  }
  if (threadIdx.x == 0) {
    int i0 = 0; float v0 = logits[0];
    for (int e = 1; e < EE; ++e) if (logits[e] > v0) { v0 = logits[e]; i0 = e; }
    int i1 = (i0 == 0) ? 1 : 0; float v1 = logits[i1];
    for (int e = 0; e < EE; ++e)
      if (e != i0 && logits[e] > v1) { v1 = logits[e]; i1 = e; }
    float p1 = __expf(v1 - v0);
    float inv = 1.0f / (1.0f + p1);
    int s0 = atomicAdd(&counts[i0], 1);
    tok_list[i0 * T + s0] = t; wt_list[i0 * T + s0] = inv;
    int s1 = atomicAdd(&counts[i1], 1);
    tok_list[i1 * T + s1] = t; wt_list[i1 * T + s1] = p1 * inv;
  }
}

__global__ void offsets_kernel(const int* __restrict__ counts, int* __restrict__ offsets) {
  if (threadIdx.x == 0 && blockIdx.x == 0) {
    int acc = 0;
    for (int e = 0; e < EE; ++e) { offsets[e] = acc; acc += counts[e]; }
  }
}

// ---------------------------------------------------------------------------
// MoE stage 1: G = silu(x@w1[e]) * (x@w3[e]) over gathered rows. w1t/w3t [E][F][D].
// ---------------------------------------------------------------------------
__launch_bounds__(256)
__global__ void moe_h_kernel(const __bf16* __restrict__ xn, const __bf16* __restrict__ w1t,
                             const __bf16* __restrict__ w3t,
                             const int* __restrict__ counts, const int* __restrict__ offsets,
                             const int* __restrict__ tok_list,
                             __bf16* __restrict__ G, int T, int D_, int F_) {
  int e = blockIdx.z;
  int cnt = counts[e];
  int bm = blockIdx.y * 128;
  if (bm >= cnt) return;
  int bn = blockIdx.x * 64;
  int tid = threadIdx.x, wv = tid >> 5, lane = tid & 31;
  int hi = (lane >> 4) & 1, ln = lane & 15;
  __shared__ __align__(16) __bf16 sA[128 * 32];
  __shared__ __align__(16) __bf16 sB1[64 * 32];
  __shared__ __align__(16) __bf16 sB3[64 * 32];
  __shared__ int srow[128];
  for (int i = tid; i < 128; i += 256) {
    int idx = bm + i; if (idx > cnt - 1) idx = cnt - 1;
    srow[i] = tok_list[e * T + idx];
  }
  __syncthreads();
  v8f acc1[4] = {}, acc3[4] = {};
  const __bf16* w1e = w1t + (size_t)e * F_ * D_;
  const __bf16* w3e = w3t + (size_t)e * F_ * D_;
  for (int k0 = 0; k0 < D_; k0 += 32) {
#ifdef HAVE_TDM
    if (wv == 0) {
      tdm_load_2d(w1e + (size_t)bn * D_ + k0, lds_off_of(sB1),
                  (unsigned)D_, (unsigned)F_, 32u, 64u, (unsigned)D_);
      tdm_load_2d(w3e + (size_t)bn * D_ + k0, lds_off_of(sB3),
                  (unsigned)D_, (unsigned)F_, 32u, 64u, (unsigned)D_);
    }
#else
    for (int i = tid; i < 256; i += 256) {
      int r = i >> 2, c = (i & 3) << 3;
      ((uint4*)sB1)[i] = *(const uint4*)(w1e + (size_t)(bn + r) * D_ + k0 + c);
      ((uint4*)sB3)[i] = *(const uint4*)(w3e + (size_t)(bn + r) * D_ + k0 + c);
    }
#endif
    {   // gathered A: two-phase
      int i0 = tid, i1 = tid + 256;
      const uint4 a0 = *(const uint4*)(xn + (size_t)srow[i0 >> 2] * D_ + k0 + ((i0 & 3) << 3));
      const uint4 a1 = *(const uint4*)(xn + (size_t)srow[i1 >> 2] * D_ + k0 + ((i1 & 3) << 3));
      ((uint4*)sA)[i0] = a0;
      ((uint4*)sA)[i1] = a1;
    }
#ifdef HAVE_TDM
    if (wv == 0) __builtin_amdgcn_s_wait_tensorcnt(0);
#endif
    __syncthreads();
    v16bf af = lds_frag(sA, wv * 16, 32, 0);
#pragma unroll
    for (int nt = 0; nt < 4; ++nt) {
      v16bf b1 = lds_frag(sB1, nt * 16, 32, 0);
      acc1[nt] = WMMA_BF(af, b1, acc1[nt]);
      v16bf b3 = lds_frag(sB3, nt * 16, 32, 0);
      acc3[nt] = WMMA_BF(af, b3, acc3[nt]);
    }
    __syncthreads();
  }
  int base = offsets[e];
#pragma unroll
  for (int nt = 0; nt < 4; ++nt) {
#pragma unroll
    for (int r = 0; r < 8; ++r) {
      int mi = bm + wv * 16 + r + hi * 8;
      if (mi < cnt) {
        float h1 = acc1[nt][r];
        float g = (h1 / (1.0f + __expf(-h1))) * acc3[nt][r];
        G[(size_t)(base + mi) * F_ + bn + nt * 16 + ln] = f2bf(g);
      }
    }
  }
}

// ---------------------------------------------------------------------------
// MoE stage 2: moe_out[token] += wt * (G @ w2[e]); w2t [E][D][F].
// ---------------------------------------------------------------------------
__launch_bounds__(256)
__global__ void moe_out_kernel(const __bf16* __restrict__ G, const __bf16* __restrict__ w2t,
                               const int* __restrict__ counts, const int* __restrict__ offsets,
                               const int* __restrict__ tok_list, const float* __restrict__ wt_list,
                               float* __restrict__ moe_out, int T, int F_, int D_) {
  int e = blockIdx.z;
  int cnt = counts[e];
  int bm = blockIdx.y * 128;
  if (bm >= cnt) return;
  int bn = blockIdx.x * 64;
  int tid = threadIdx.x, wv = tid >> 5, lane = tid & 31;
  int hi = (lane >> 4) & 1, ln = lane & 15;
  __shared__ __align__(16) __bf16 sA[128 * 32];
  __shared__ __align__(16) __bf16 sB[64 * 32];
  int base = offsets[e];
  const __bf16* w2e = w2t + (size_t)e * D_ * F_;
  v8f acc[4] = {};
  for (int k0 = 0; k0 < F_; k0 += 32) {
#ifdef HAVE_TDM
    if (wv == 0)
      tdm_load_2d(w2e + (size_t)bn * F_ + k0, lds_off_of(sB),
                  (unsigned)F_, (unsigned)D_, 32u, 64u, (unsigned)F_);
#else
    for (int i = tid; i < 256; i += 256) {
      int r = i >> 2, c = (i & 3) << 3;
      ((uint4*)sB)[i] = *(const uint4*)(w2e + (size_t)(bn + r) * F_ + k0 + c);
    }
#endif
    {
      int i0 = tid, i1 = tid + 256;
      int mr0 = bm + (i0 >> 2); if (mr0 > cnt - 1) mr0 = cnt - 1;
      int mr1 = bm + (i1 >> 2); if (mr1 > cnt - 1) mr1 = cnt - 1;
      const uint4 a0 = *(const uint4*)(G + (size_t)(base + mr0) * F_ + k0 + ((i0 & 3) << 3));
      const uint4 a1 = *(const uint4*)(G + (size_t)(base + mr1) * F_ + k0 + ((i1 & 3) << 3));
      ((uint4*)sA)[i0] = a0;
      ((uint4*)sA)[i1] = a1;
    }
#ifdef HAVE_TDM
    if (wv == 0) __builtin_amdgcn_s_wait_tensorcnt(0);
#endif
    __syncthreads();
    v16bf af = lds_frag(sA, wv * 16, 32, 0);
#pragma unroll
    for (int nt = 0; nt < 4; ++nt) {
      v16bf bf = lds_frag(sB, nt * 16, 32, 0);
      acc[nt] = WMMA_BF(af, bf, acc[nt]);
    }
    __syncthreads();
  }
#pragma unroll
  for (int nt = 0; nt < 4; ++nt) {
#pragma unroll
    for (int r = 0; r < 8; ++r) {
      int mi = bm + wv * 16 + r + hi * 8;
      if (mi < cnt) {
        int tok = tok_list[e * T + mi];
        float w = wt_list[e * T + mi];
        atomicAdd(&moe_out[(size_t)tok * D_ + bn + nt * 16 + ln], w * acc[nt][r]);
      }
    }
  }
}

// ---------------------------------------------------------------------------
// Host orchestration
// ---------------------------------------------------------------------------
static inline char* carve(char*& p, size_t bytes) {
  char* r = p;
  p += (bytes + 255) & ~(size_t)255;
  return r;
}

extern "C" void kernel_launch(void* const* d_in, const int* in_sizes, int n_in,
                              void* d_out, int out_size, void* d_ws, size_t ws_size,
                              hipStream_t stream) {
  (void)in_sizes; (void)n_in; (void)out_size; (void)ws_size;
  const float* x      = (const float*)d_in[0];
  const int*   xpos   = (const int*)d_in[1];
  const float* ln1_w  = (const float*)d_in[2];
  const float* ln2_w  = (const float*)d_in[3];
  const float* wq     = (const float*)d_in[4];
  const float* wk     = (const float*)d_in[5];
  const float* wv     = (const float*)d_in[6];
  const float* wo     = (const float*)d_in[7];
  const float* gate_w = (const float*)d_in[8];
  const float* w1     = (const float*)d_in[9];
  const float* w3     = (const float*)d_in[10];
  const float* w2     = (const float*)d_in[11];

  char* p = (char*)d_ws;
  __bf16* wq_t   = (__bf16*)carve(p, (size_t)DD * DD * 2);
  __bf16* wk_t   = (__bf16*)carve(p, (size_t)DD * DD * 2);
  __bf16* wv_t   = (__bf16*)carve(p, (size_t)DD * DD * 2);
  __bf16* wo_t   = (__bf16*)carve(p, (size_t)DD * DD * 2);
  __bf16* w1_t   = (__bf16*)carve(p, (size_t)EE * DD * FF * 2);
  __bf16* w3_t   = (__bf16*)carve(p, (size_t)EE * DD * FF * 2);
  __bf16* w2_t   = (__bf16*)carve(p, (size_t)EE * FF * DD * 2);
  __bf16* xn1    = (__bf16*)carve(p, (size_t)TT * DD * 2);
  __bf16* qbf    = (__bf16*)carve(p, (size_t)TT * DD * 2);
  __bf16* kbf    = (__bf16*)carve(p, (size_t)TT * DD * 2);
  __bf16* vT     = (__bf16*)carve(p, (size_t)TT * DD * 2);   // [B][H][HD][S]
  __bf16* attnbf = (__bf16*)carve(p, (size_t)TT * DD * 2);
  float*  tmpf   = (float*)carve(p, (size_t)TT * DD * 4);
  float*  hbuf   = (float*)carve(p, (size_t)TT * DD * 4);
  __bf16* xn2    = (__bf16*)carve(p, (size_t)TT * DD * 2);
  float*  moebuf = (float*)carve(p, (size_t)TT * DD * 4);
  int*    counts = (int*)carve(p, EE * sizeof(int));
  int*    offs   = (int*)carve(p, EE * sizeof(int));
  int*    toks   = (int*)carve(p, (size_t)EE * TT * sizeof(int));
  float*  wts    = (float*)carve(p, (size_t)EE * TT * sizeof(float));
  __bf16* Gbuf   = (__bf16*)carve(p, (size_t)(2 * TT) * FF * 2);

  cvt_t_bf16_kernel<<<dim3(DD / 32, DD / 32, 1), 256, 0, stream>>>(wq, wq_t, DD, DD);
  cvt_t_bf16_kernel<<<dim3(DD / 32, DD / 32, 1), 256, 0, stream>>>(wk, wk_t, DD, DD);
  cvt_t_bf16_kernel<<<dim3(DD / 32, DD / 32, 1), 256, 0, stream>>>(wv, wv_t, DD, DD);
  cvt_t_bf16_kernel<<<dim3(DD / 32, DD / 32, 1), 256, 0, stream>>>(wo, wo_t, DD, DD);
  cvt_t_bf16_kernel<<<dim3(FF / 32, DD / 32, EE), 256, 0, stream>>>(w1, w1_t, DD, FF);
  cvt_t_bf16_kernel<<<dim3(FF / 32, DD / 32, EE), 256, 0, stream>>>(w3, w3_t, DD, FF);
  cvt_t_bf16_kernel<<<dim3(DD / 32, FF / 32, EE), 256, 0, stream>>>(w2, w2_t, FF, DD);

  rmsnorm_kernel<<<TT, 256, 0, stream>>>(x, ln1_w, xn1, DD);

  dim3 gproj(DD / 64, TT / 128);
  gemm_bf16_kernel<<<gproj, 256, 0, stream>>>(xn1, wq_t, qbf, TT, DD, DD, 1);
  gemm_bf16_kernel<<<gproj, 256, 0, stream>>>(xn1, wk_t, kbf, TT, DD, DD, 1);
  gemm_bf16_kernel<<<gproj, 256, 0, stream>>>(xn1, wv_t, vT, TT, DD, DD, 2);

  int rope_threads = TT * HH * (HDHD / 2);
  rope_kernel<<<(rope_threads + 255) / 256, 256, 0, stream>>>(qbf, kbf, xpos, TT, HH, HDHD);

  attn_kernel<<<dim3(BB * HH, SS / 64), 128, 0, stream>>>(qbf, kbf, vT, attnbf, SS, HH, HDHD);

  gemm_bf16_kernel<<<gproj, 256, 0, stream>>>(attnbf, wo_t, tmpf, TT, DD, DD, 0);
  size_t nTD = (size_t)TT * DD;
  add2_kernel<<<(unsigned)((nTD + 255) / 256), 256, 0, stream>>>(x, tmpf, hbuf, nTD);

  rmsnorm_kernel<<<TT, 256, 0, stream>>>(hbuf, ln2_w, xn2, DD);
  (void)hipMemsetAsync(counts, 0, EE * sizeof(int), stream);
  (void)hipMemsetAsync(moebuf, 0, nTD * sizeof(float), stream);
  gate_topk_kernel<<<TT, 256, 0, stream>>>(xn2, gate_w, counts, toks, wts, TT, DD);
  offsets_kernel<<<1, 32, 0, stream>>>(counts, offs);

  moe_h_kernel<<<dim3(FF / 64, TT / 128, EE), 256, 0, stream>>>(
      xn2, w1_t, w3_t, counts, offs, toks, Gbuf, TT, DD, FF);
  moe_out_kernel<<<dim3(DD / 64, TT / 128, EE), 256, 0, stream>>>(
      Gbuf, w2_t, counts, offs, toks, wts, moebuf, TT, FF, DD);

  add2_kernel<<<(unsigned)((nTD + 255) / 256), 256, 0, stream>>>(hbuf, moebuf, (float*)d_out, nTD);
}
